// Net_75127567941717
// MI455X (gfx1250) — matrix operand order repaired and checked by reading, not compiled, not run on hardware
//
#include <hip/hip_runtime.h>
#include <hip/hip_bf16.h>
#include <math.h>

typedef __attribute__((ext_vector_type(16))) _Float16 v16h;
typedef __attribute__((ext_vector_type(8)))  float    v8f;

#define B_      512
#define N_      128
#define K_      20
#define F_      16
#define EDGES_  (B_ * N_ * K_)   // 1,310,720
#define EGROUPS (EDGES_ / 16)    // 81,920 wave groups
#define EBLOCKS (EGROUPS / 8)    // 10,240 blocks of 8 waves

#define LD1     40               // padded row stride (halves) for 32-wide tiles
#define LD2     72               // padded row stride (halves) for 64-wide tiles

// ---------------------------------------------------------------------------
// WMMA fragment loaders (CDNA5 16x16x32 f16 layouts, ISA 7.12.2)
// Both loaders read two aligned 16-byte contiguous groups per lane.
// ---------------------------------------------------------------------------
__device__ inline v16h frag_a(const _Float16* buf, int ld, int k0, int lane) {
  // A 16x32 f16: lane L holds row (L&15); element e -> K = (e&7)+16*(e>>3)+8*(L>>4)
  const _Float16* p = buf + (lane & 15) * ld + ((lane >> 4) << 3) + k0;
  v16h a;
#pragma unroll
  for (int e = 0; e < 8; ++e) a[e] = p[e];
#pragma unroll
  for (int e = 0; e < 8; ++e) a[8 + e] = p[16 + e];
  return a;
}

__device__ inline v16h frag_bt(const _Float16* wt, int ldk, int n0, int k0, int lane) {
  // B 32x16 f16 from TRANSPOSED weight staging wt[n*ldk + k]:
  // lane L holds col (L&15); element e -> K = k0 + (e&7)+16*(e>>3)+8*(L>>4)
  const _Float16* p = wt + (n0 + (lane & 15)) * ldk + k0 + ((lane >> 4) << 3);
  v16h b;
#pragma unroll
  for (int e = 0; e < 8; ++e) b[e] = p[e];
#pragma unroll
  for (int e = 0; e < 8; ++e) b[8 + e] = p[16 + e];
  return b;
}

__device__ inline void wavefence() { __builtin_amdgcn_wave_barrier(); }

// ---------------------------------------------------------------------------
// Utility kernels
// ---------------------------------------------------------------------------
__global__ void zero_kernel(float* p, int n) {
  int i = blockIdx.x * blockDim.x + threadIdx.x;
  if (i < n) p[i] = 0.0f;
}

__global__ void finalize_kernel(const float* S, const float* Q, float* M, float* I,
                                int C, float cnt) {
  int c = threadIdx.x;
  if (c < C) {
    float mu  = S[c] / cnt;
    float var = Q[c] / cnt - mu * mu;
    M[c] = mu;
    I[c] = rsqrtf(var + 1e-5f);
  }
}

// ---------------------------------------------------------------------------
// KNN: one block per batch, 128 threads (one per point); top-20 by insertion.
// Distance matches reference: |i|^2 + |j|^2 - 2 i.j  (+1e9 on diagonal).
// ---------------------------------------------------------------------------
template <int D>
__global__ __launch_bounds__(128) void knn_kernel(const float* __restrict__ pts,
                                                  int* __restrict__ idx) {
  __shared__ float P[128 * D];
  __shared__ float SQ[128];
  int b = blockIdx.x, t = threadIdx.x;
  for (int i = t; i < 128 * D; i += 128) P[i] = pts[(long)b * 128 * D + i];
  __syncthreads();
  float s = 0.0f;
#pragma unroll
  for (int c = 0; c < D; ++c) { float v = P[t * D + c]; s += v * v; }
  SQ[t] = s;
  __syncthreads();
  float pr[D];
#pragma unroll
  for (int c = 0; c < D; ++c) pr[c] = P[t * D + c];
  float bd[K_]; int bi[K_];
#pragma unroll
  for (int q = 0; q < K_; ++q) { bd[q] = 3.0e38f; bi[q] = 0; }
  for (int mI = 0; mI < 128; ++mI) {
    float dot = 0.0f;
#pragma unroll
    for (int c = 0; c < D; ++c) dot += pr[c] * P[mI * D + c];
    float d = s + SQ[mI] - 2.0f * dot;
    if (mI == t) d += 1.0e9f;
    if (d < bd[K_ - 1]) {
      int q = K_ - 1;
      while (q > 0 && bd[q - 1] > d) { bd[q] = bd[q - 1]; bi[q] = bi[q - 1]; --q; }
      bd[q] = d; bi[q] = mI;
    }
  }
  for (int q = 0; q < K_; ++q) idx[((long)b * 128 + t) * K_ + q] = bi[q];
}

// ---------------------------------------------------------------------------
// EdgeConv1: 32-d edge vector -> 32 -> 32 -> 32. STAGE 1/2 gather BN stats,
// STAGE 3 produces x1 (mean over K via atomics).
// ---------------------------------------------------------------------------
template <int STAGE>
__global__ __launch_bounds__(256) void ec1_kernel(
    const float* __restrict__ feat, const int* __restrict__ idx,
    const float* __restrict__ w1, const float* __restrict__ b1v,
    const float* __restrict__ g1, const float* __restrict__ be1,
    const float* __restrict__ w2, const float* __restrict__ b2v,
    const float* __restrict__ g2, const float* __restrict__ be2,
    const float* __restrict__ w3, const float* __restrict__ b3v,
    const float* __restrict__ mean1, const float* __restrict__ istd1,
    const float* __restrict__ mean2, const float* __restrict__ istd2,
    float* __restrict__ statS, float* __restrict__ statQ,
    float* __restrict__ xout) {
  __shared__ _Float16 sW1[32 * LD1], sW2[32 * LD1], sW3[32 * LD1];
  __shared__ _Float16 sA[8][16 * LD1], sH[8][16 * LD1];
  __shared__ float sS[32], sQ[32];
  int tid = threadIdx.x;
  for (int i = tid; i < 1024; i += 256) {
    int k = i >> 5, n = i & 31;           // w is [k][n] row-major -> store [n][k]
    sW1[n * LD1 + k] = (_Float16)w1[i];
    if (STAGE >= 2) sW2[n * LD1 + k] = (_Float16)w2[i];
    if (STAGE >= 3) sW3[n * LD1 + k] = (_Float16)w3[i];
  }
  if (STAGE <= 2 && tid < 32) { sS[tid] = 0.0f; sQ[tid] = 0.0f; }
  __syncthreads();

  int wv = tid >> 5, lane = tid & 31;
  int gw = blockIdx.x * 8 + wv;
  int eb = gw * 16;
  int m = lane & 15, p = lane >> 4;
  int e = eb + m;
  int b  = e / (N_ * K_);
  int rm = e - b * (N_ * K_);
  int n  = rm / K_;
  const float* xi = feat + ((long)b * N_ + n) * F_;
  int j = idx[e];
  const float* xj = feat + ((long)b * N_ + j) * F_;
  _Float16* A = sA[wv];
  _Float16* H = sH[wv];
  if (p == 0) {
#pragma unroll
    for (int f = 0; f < 16; ++f) A[m * LD1 + f] = (_Float16)xi[f];
  } else {
#pragma unroll
    for (int f = 0; f < 16; ++f) A[m * LD1 + 16 + f] = (_Float16)(xj[f] - xi[f]);
  }
  wavefence();

  int nlo = lane & 15;
  int rbase = (lane >> 4) << 3;

  // ---- layer 1 ----
  v16h af = frag_a(A, LD1, 0, lane);
#pragma unroll
  for (int t = 0; t < 2; ++t) {
    int nc = t * 16 + nlo;
    v16h bf = frag_bt(sW1, LD1, t * 16, 0, lane);
    v8f d;
    float bb = b1v[nc];
#pragma unroll
    for (int r = 0; r < 8; ++r) d[r] = bb;
    d = __builtin_amdgcn_wmma_f32_16x16x32_f16(false, af, false, bf, (short)0, d, false, false);
    if (STAGE == 1) {
      float s = 0.0f, q = 0.0f;
#pragma unroll
      for (int r = 0; r < 8; ++r) { s += d[r]; q += d[r] * d[r]; }
      atomicAdd(&sS[nc], s); atomicAdd(&sQ[nc], q);
    } else {
      float mu = mean1[nc], is = istd1[nc], ga = g1[nc], bt = be1[nc];
#pragma unroll
      for (int r = 0; r < 8; ++r) {
        float v = (d[r] - mu) * is * ga + bt;
        H[(rbase + r) * LD1 + nc] = (_Float16)fmaxf(v, 0.0f);
      }
    }
  }

  // ---- layer 2 ----
  if (STAGE >= 2) {
    wavefence();
    v16h af2 = frag_a(H, LD1, 0, lane);
#pragma unroll
    for (int t = 0; t < 2; ++t) {
      int nc = t * 16 + nlo;
      v16h bf = frag_bt(sW2, LD1, t * 16, 0, lane);
      v8f d;
      float bb = b2v[nc];
#pragma unroll
      for (int r = 0; r < 8; ++r) d[r] = bb;
      d = __builtin_amdgcn_wmma_f32_16x16x32_f16(false, af2, false, bf, (short)0, d, false, false);
      if (STAGE == 2) {
        float s = 0.0f, q = 0.0f;
#pragma unroll
        for (int r = 0; r < 8; ++r) { s += d[r]; q += d[r] * d[r]; }
        atomicAdd(&sS[nc], s); atomicAdd(&sQ[nc], q);
      } else {
        float mu = mean2[nc], is = istd2[nc], ga = g2[nc], bt = be2[nc];
#pragma unroll
        for (int r = 0; r < 8; ++r) {
          float v = (d[r] - mu) * is * ga + bt;
          A[(rbase + r) * LD1 + nc] = (_Float16)fmaxf(v, 0.0f);  // reuse A buffer
        }
      }
    }
  }

  // ---- layer 3 + mean over K ----
  if (STAGE >= 3) {
    wavefence();
    v16h af3 = frag_a(A, LD1, 0, lane);
#pragma unroll
    for (int t = 0; t < 2; ++t) {
      int nc = t * 16 + nlo;
      v16h bf = frag_bt(sW3, LD1, t * 16, 0, lane);
      v8f d;
      float bb = b3v[nc];
#pragma unroll
      for (int r = 0; r < 8; ++r) d[r] = bb;
      d = __builtin_amdgcn_wmma_f32_16x16x32_f16(false, af3, false, bf, (short)0, d, false, false);
#pragma unroll
      for (int r = 0; r < 8; ++r) {
        int mm = rbase + r;
        int ee = eb + mm;
        int bb2 = ee / (N_ * K_);
        int rr  = ee - bb2 * (N_ * K_);
        int nn  = rr / K_;
        atomicAdd(&xout[((long)bb2 * N_ + nn) * 32 + nc], d[r] * (1.0f / 20.0f));
      }
    }
  }

  if (STAGE <= 2) {
    __syncthreads();
    if (tid < 32) { atomicAdd(&statS[tid], sS[tid]); atomicAdd(&statQ[tid], sQ[tid]); }
  }
}

// ---------------------------------------------------------------------------
// EdgeConv2: 64-d edge vector -> 64 -> 64 (two K-chunks per WMMA tile).
// ---------------------------------------------------------------------------
template <int STAGE>
__global__ __launch_bounds__(256) void ec2_kernel(
    const float* __restrict__ xin, const int* __restrict__ idx,
    const float* __restrict__ w1, const float* __restrict__ b1v,
    const float* __restrict__ g1, const float* __restrict__ be1,
    const float* __restrict__ w2, const float* __restrict__ b2v,
    const float* __restrict__ mean1, const float* __restrict__ istd1,
    float* __restrict__ statS, float* __restrict__ statQ,
    float* __restrict__ xout) {
  __shared__ _Float16 sW1[64 * LD2], sW2[64 * LD2];
  __shared__ _Float16 sA[8][16 * LD2], sH[8][16 * LD2];
  __shared__ float sS[64], sQ[64];
  int tid = threadIdx.x;
  for (int i = tid; i < 4096; i += 256) {
    int k = i >> 6, n = i & 63;           // store transposed [n][k]
    sW1[n * LD2 + k] = (_Float16)w1[i];
    if (STAGE >= 2) sW2[n * LD2 + k] = (_Float16)w2[i];
  }
  if (STAGE == 1 && tid < 64) { sS[tid] = 0.0f; sQ[tid] = 0.0f; }
  __syncthreads();

  int wv = tid >> 5, lane = tid & 31;
  int gw = blockIdx.x * 8 + wv;
  int eb = gw * 16;
  int m = lane & 15, p = lane >> 4;
  int e = eb + m;
  int b  = e / (N_ * K_);
  int rm = e - b * (N_ * K_);
  int n  = rm / K_;
  const float* xi = xin + ((long)b * N_ + n) * 32;
  int j = idx[e];
  const float* xj = xin + ((long)b * N_ + j) * 32;
  _Float16* A = sA[wv];
  _Float16* H = sH[wv];
  if (p == 0) {
#pragma unroll
    for (int f = 0; f < 32; ++f) A[m * LD2 + f] = (_Float16)xi[f];
  } else {
#pragma unroll
    for (int f = 0; f < 32; ++f) A[m * LD2 + 32 + f] = (_Float16)(xj[f] - xi[f]);
  }
  wavefence();

  int nlo = lane & 15;
  int rbase = (lane >> 4) << 3;

  // ---- layer 1 (64 -> 64) ----
  v16h a0 = frag_a(A, LD2, 0, lane);
  v16h a1 = frag_a(A, LD2, 32, lane);
#pragma unroll
  for (int t = 0; t < 4; ++t) {
    int nc = t * 16 + nlo;
    v8f d;
    float bb = b1v[nc];
#pragma unroll
    for (int r = 0; r < 8; ++r) d[r] = bb;
    v16h bf0 = frag_bt(sW1, LD2, t * 16, 0, lane);
    d = __builtin_amdgcn_wmma_f32_16x16x32_f16(false, a0, false, bf0, (short)0, d, false, false);
    v16h bf1 = frag_bt(sW1, LD2, t * 16, 32, lane);
    d = __builtin_amdgcn_wmma_f32_16x16x32_f16(false, a1, false, bf1, (short)0, d, false, false);
    if (STAGE == 1) {
      float s = 0.0f, q = 0.0f;
#pragma unroll
      for (int r = 0; r < 8; ++r) { s += d[r]; q += d[r] * d[r]; }
      atomicAdd(&sS[nc], s); atomicAdd(&sQ[nc], q);
    } else {
      float mu = mean1[nc], is = istd1[nc], ga = g1[nc], bt = be1[nc];
#pragma unroll
      for (int r = 0; r < 8; ++r) {
        float v = (d[r] - mu) * is * ga + bt;
        H[(rbase + r) * LD2 + nc] = (_Float16)fmaxf(v, 0.0f);
      }
    }
  }

  // ---- layer 2 (64 -> 64) + mean over K ----
  if (STAGE >= 2) {
    wavefence();
    v16h h0 = frag_a(H, LD2, 0, lane);
    v16h h1 = frag_a(H, LD2, 32, lane);
#pragma unroll
    for (int t = 0; t < 4; ++t) {
      int nc = t * 16 + nlo;
      v8f d;
      float bb = b2v[nc];
#pragma unroll
      for (int r = 0; r < 8; ++r) d[r] = bb;
      v16h bf0 = frag_bt(sW2, LD2, t * 16, 0, lane);
      d = __builtin_amdgcn_wmma_f32_16x16x32_f16(false, h0, false, bf0, (short)0, d, false, false);
      v16h bf1 = frag_bt(sW2, LD2, t * 16, 32, lane);
      d = __builtin_amdgcn_wmma_f32_16x16x32_f16(false, h1, false, bf1, (short)0, d, false, false);
#pragma unroll
      for (int r = 0; r < 8; ++r) {
        int mm = rbase + r;
        int ee = eb + mm;
        int bb2 = ee / (N_ * K_);
        int rr  = ee - bb2 * (N_ * K_);
        int nn  = rr / K_;
        atomicAdd(&xout[((long)bb2 * N_ + nn) * 64 + nc], d[r] * (1.0f / 20.0f));
      }
    }
  }

  if (STAGE == 1) {
    __syncthreads();
    if (tid < 64) { atomicAdd(&statS[tid], sS[tid]); atomicAdd(&statQ[tid], sQ[tid]); }
  }
}

// ---------------------------------------------------------------------------
// Max over N, then head MLP 64->128->128->1
// ---------------------------------------------------------------------------
__global__ __launch_bounds__(64) void maxpool_kernel(const float* __restrict__ x2,
                                                     float* __restrict__ g) {
  int b = blockIdx.x, c = threadIdx.x;
  float mx = -3.0e38f;
  for (int n = 0; n < N_; ++n) mx = fmaxf(mx, x2[((long)b * N_ + n) * 64 + c]);
  g[b * 64 + c] = mx;
}

__global__ __launch_bounds__(128) void head_kernel(
    const float* __restrict__ g,
    const float* __restrict__ w1, const float* __restrict__ b1,
    const float* __restrict__ w2, const float* __restrict__ b2,
    const float* __restrict__ w3, const float* __restrict__ b3,
    float* __restrict__ out) {
  __shared__ float sg[64], h1[128], h2[128], red[128];
  int b = blockIdx.x, t = threadIdx.x;
  if (t < 64) sg[t] = g[b * 64 + t];
  __syncthreads();
  float acc = b1[t];
  for (int i = 0; i < 64; ++i) acc += sg[i] * w1[i * 128 + t];
  h1[t] = fmaxf(acc, 0.0f);
  __syncthreads();
  acc = b2[t];
  for (int i = 0; i < 128; ++i) acc += h1[i] * w2[i * 128 + t];
  h2[t] = fmaxf(acc, 0.0f);
  __syncthreads();
  red[t] = h2[t] * w3[t];
  __syncthreads();
  for (int s = 64; s > 0; s >>= 1) {
    if (t < s) red[t] += red[t + s];
    __syncthreads();
  }
  if (t == 0) out[b] = red[0] + b3[0];
}

// ---------------------------------------------------------------------------
extern "C" void kernel_launch(void* const* d_in, const int* in_sizes, int n_in,
                              void* d_out, int out_size, void* d_ws, size_t ws_size,
                              hipStream_t stream) {
  const float* points = (const float*)d_in[0];
  const float* feat   = (const float*)d_in[1];
  const float* c1_w1 = (const float*)d_in[2];
  const float* c1_b1 = (const float*)d_in[3];
  const float* c1_g1 = (const float*)d_in[4];
  const float* c1_be1 = (const float*)d_in[5];
  const float* c1_w2 = (const float*)d_in[6];
  const float* c1_b2 = (const float*)d_in[7];
  const float* c1_g2 = (const float*)d_in[8];
  const float* c1_be2 = (const float*)d_in[9];
  const float* c1_w3 = (const float*)d_in[10];
  const float* c1_b3 = (const float*)d_in[11];
  const float* c2_w1 = (const float*)d_in[12];
  const float* c2_b1 = (const float*)d_in[13];
  const float* c2_g1 = (const float*)d_in[14];
  const float* c2_be1 = (const float*)d_in[15];
  const float* c2_w2 = (const float*)d_in[16];
  const float* c2_b2 = (const float*)d_in[17];
  const float* m_w1 = (const float*)d_in[18];
  const float* m_b1 = (const float*)d_in[19];
  const float* m_w2 = (const float*)d_in[20];
  const float* m_b2 = (const float*)d_in[21];
  const float* m_w3 = (const float*)d_in[22];
  const float* m_b3 = (const float*)d_in[23];
  float* out = (float*)d_out;

  char* ws = (char*)d_ws;
  float* st = (float*)ws;
  float *gS1 = st + 0,   *gQ1 = st + 32,  *gM1 = st + 64,  *gI1 = st + 96;
  float *gS2 = st + 128, *gQ2 = st + 160, *gM2 = st + 192, *gI2 = st + 224;
  float *gS3 = st + 256, *gQ3 = st + 320, *gM3 = st + 384, *gI3 = st + 448;
  int* idx1 = (int*)(ws + 4096);
  int* idx2 = idx1 + (size_t)EDGES_;
  float* x1 = (float*)(idx2 + (size_t)EDGES_);
  float* x2 = x1 + (size_t)B_ * N_ * 32;
  float* g  = x2 + (size_t)B_ * N_ * 64;

  const float cnt = (float)EDGES_;
  const int NX1 = B_ * N_ * 32;   // 2,097,152
  const int NX2 = B_ * N_ * 64;   // 4,194,304

  zero_kernel<<<2, 256, 0, stream>>>(st, 512);
  zero_kernel<<<(NX1 + 255) / 256, 256, 0, stream>>>(x1, NX1);
  zero_kernel<<<(NX2 + 255) / 256, 256, 0, stream>>>(x2, NX2);

  knn_kernel<2><<<B_, 128, 0, stream>>>(points, idx1);

  ec1_kernel<1><<<EBLOCKS, 256, 0, stream>>>(feat, idx1,
      c1_w1, c1_b1, c1_g1, c1_be1, c1_w2, c1_b2, c1_g2, c1_be2, c1_w3, c1_b3,
      gM1, gI1, gM2, gI2, gS1, gQ1, x1);
  finalize_kernel<<<1, 64, 0, stream>>>(gS1, gQ1, gM1, gI1, 32, cnt);
  ec1_kernel<2><<<EBLOCKS, 256, 0, stream>>>(feat, idx1,
      c1_w1, c1_b1, c1_g1, c1_be1, c1_w2, c1_b2, c1_g2, c1_be2, c1_w3, c1_b3,
      gM1, gI1, gM2, gI2, gS2, gQ2, x1);
  finalize_kernel<<<1, 64, 0, stream>>>(gS2, gQ2, gM2, gI2, 32, cnt);
  ec1_kernel<3><<<EBLOCKS, 256, 0, stream>>>(feat, idx1,
      c1_w1, c1_b1, c1_g1, c1_be1, c1_w2, c1_b2, c1_g2, c1_be2, c1_w3, c1_b3,
      gM1, gI1, gM2, gI2, gS1, gQ1, x1);

  knn_kernel<32><<<B_, 128, 0, stream>>>(x1, idx2);

  ec2_kernel<1><<<EBLOCKS, 256, 0, stream>>>(x1, idx2,
      c2_w1, c2_b1, c2_g1, c2_be1, c2_w2, c2_b2,
      gM3, gI3, gS3, gQ3, x2);
  finalize_kernel<<<1, 64, 0, stream>>>(gS3, gQ3, gM3, gI3, 64, cnt);
  ec2_kernel<2><<<EBLOCKS, 256, 0, stream>>>(x1, idx2,
      c2_w1, c2_b1, c2_g1, c2_be1, c2_w2, c2_b2,
      gM3, gI3, gS3, gQ3, x2);

  maxpool_kernel<<<B_, 64, 0, stream>>>(x2, g);
  head_kernel<<<B_, 128, 0, stream>>>(g, m_w1, m_b1, m_w2, m_b2, m_w3, m_b3, out);
}
